// AGMBrain_11991548690849
// MI455X (gfx1250) — matrix operand order repaired and checked by reading, not compiled, not run on hardware
//
#include <hip/hip_runtime.h>
#include <hip/hip_bf16.h>

// ---------------------------------------------------------------------------
// AGMBrain on MI455X (gfx1250, wave32, WMMA)
//
// Core op per step: messages[bd,i] = sum_{j,n} E[i,j,n] * s[bd,j] * s[bd,n]
//   == GEMM (4096 x 16384) x (16384 x 128) with A[bd,(j,n)] = s[bd,j]*s[bd,n]
// A synthesized in-register from LDS-staged state rows; B is a bf16 copy of
// edge_vectors laid out [i][k] (L2-resident, 4MB). f32 accumulate via
// v_wmma_f32_16x16x32_bf16.
//
// v4: each wave owns FOUR i-tiles: one A build (16 packed VALU) feeds four
// back-to-back WMMAs (no D->A/B hazards between them), amortizing the A-build
// VALU, the ds_load burst and the trailing WMMA->VALU hazard fill 4x.
// Block = 4 waves covering 32 bd-rows x 128 i-cols; the 4 B-tile rows are
// 512KB apart -> single base address + constant 24-bit instruction offsets.
// 2-stage ping-pong software pipeline keeps B loads one chunk ahead.
//
// Workspace (needs 8 MB):
//   [0,2MB)   statesA  f32 [4096][128]
//   [2,4MB)   statesB  f32 [4096][128]
//   [4,8MB)   Ebf      bf16 [128][16384]
// ---------------------------------------------------------------------------

typedef __attribute__((ext_vector_type(16))) __bf16 bf16x16;
typedef __attribute__((ext_vector_type(8)))  float  f32x8;

#define NN 128
#define BB 32
#define INF 4608
#define BD (BB * NN)          // 4096 rows
#define KTOT (NN * NN)        // 16384 reduction length
#define NCHUNK (KTOT / 32)    // 512 K-chunks of 32
#define TS (16 * KTOT)        // element stride between i-tiles in Ebf
#define SROW 132              // padded LDS row stride (floats): 16B-aligned
                              // float4 loads, conflict-free banks

#define WMMA_BF16(A, B, C) \
    __builtin_amdgcn_wmma_f32_16x16x32_bf16(false, (A), false, (B), (short)0, (C), false, false)

// -------------------------- 1) input proj + state init ---------------------
__global__ void proj_init_kernel(const float* __restrict__ x,
                                 const float* __restrict__ W,
                                 const float* __restrict__ bias,
                                 const float* __restrict__ neuron,
                                 float* __restrict__ states0) {
    __shared__ float xs[128];
    __shared__ float xt[128];
    const int b = blockIdx.x;      // 0..31
    const int d = threadIdx.x;     // 0..127
    float acc = bias[d];
    const float* wrow = W + (size_t)d * INF;
    for (int k0 = 0; k0 < INF; k0 += 128) {
        __syncthreads();
        xs[d] = x[(size_t)b * INF + k0 + d];
        __syncthreads();
        #pragma unroll 8
        for (int t = 0; t < 128; ++t) acc += xs[t] * wrow[k0 + t];
    }
    xt[d] = acc;
    __syncthreads();
    // states0[(b*128+dd)*128+n] = neuron[n,dd] + xt[dd]; flat = b*16384 + e
    for (int e = d; e < NN * NN; e += 128) {
        int dd = e >> 7, n = e & 127;
        states0[(size_t)b * (NN * NN) + e] = neuron[n * NN + dd] + xt[dd];
    }
}

// -------------------------- 2) edge_vectors -> bf16 ------------------------
// edge flat [i][j][n] == Ebf flat [i][k] with k=j*128+n: pure convert.
__global__ void conv_e_kernel(const float* __restrict__ edge,
                              __bf16* __restrict__ ebf) {
    size_t idx = (size_t)blockIdx.x * blockDim.x + threadIdx.x;
    const size_t total = (size_t)NN * NN * NN;
    for (; idx < total; idx += (size_t)gridDim.x * blockDim.x)
        ebf[idx] = (__bf16)edge[idx];
}

// -------------------------- 3) one message-passing step --------------------
// A-frag (16-bit A 16x32): lane m=L&15, kg=L>>4 holds k-offsets
// {16kg..16kg+15} (elems 0..15), matching B's contiguous order.
__device__ __forceinline__ bf16x16 build_a(const float* __restrict__ srow,
                                           int c, int kg) {
    const int j     = c >> 2;                    // chunk -> (j, n-base)
    const int nbase = (c & 3) << 5;
    const float sj  = srow[j];
    const int n0    = nbase + (kg << 4);
    float4 f0 = *(const float4*)(srow + n0);
    float4 f1 = *(const float4*)(srow + n0 + 4);
    float4 f2 = *(const float4*)(srow + n0 + 8);
    float4 f3 = *(const float4*)(srow + n0 + 12);
    bf16x16 a;
    a[0] = (__bf16)(sj * f0.x);  a[1] = (__bf16)(sj * f0.y);
    a[2] = (__bf16)(sj * f0.z);  a[3] = (__bf16)(sj * f0.w);
    a[4] = (__bf16)(sj * f1.x);  a[5] = (__bf16)(sj * f1.y);
    a[6] = (__bf16)(sj * f1.z);  a[7] = (__bf16)(sj * f1.w);
    a[8] = (__bf16)(sj * f2.x);  a[9] = (__bf16)(sj * f2.y);
    a[10] = (__bf16)(sj * f2.z); a[11] = (__bf16)(sj * f2.w);
    a[12] = (__bf16)(sj * f3.x); a[13] = (__bf16)(sj * f3.y);
    a[14] = (__bf16)(sj * f3.z); a[15] = (__bf16)(sj * f3.w);
    return a;
}

// B-frag: lane holds col (L&15), K = 16*kg + e -> one contiguous aligned 32B.
// Tile t offset (t*TS) is a compile-time constant folded into the 24-bit
// instruction offset (t*512KB < 8MB).
__device__ __forceinline__ bf16x16 load_b(const __bf16* __restrict__ erow,
                                          int c, int kg, int t) {
    return *(const bf16x16*)(erow + (size_t)t * TS + (c << 5) + (kg << 4));
}

__global__ void step_kernel(const float* __restrict__ sin_,
                            const __bf16* __restrict__ ebf,
                            float* __restrict__ sout) {
    __shared__ float sm[32 * SROW];              // 32 state rows, padded
    const int tid = threadIdx.x;                 // 128 threads = 4 waves
    const int rowBase = blockIdx.x * 32;

    // stage 32 rows of states into LDS (uniform trip count, EXEC all-1)
    for (int e = tid; e < 32 * NN; e += 128) {
        int mr = e >> 7, n = e & 127;
        sm[mr * SROW + n] = sin_[(size_t)(rowBase + mr) * NN + n];
    }
    __syncthreads();

    const int lane   = tid & 31;
    const int wv     = tid >> 5;                 // 4 waves
    const int mGroup = wv >> 1;                  // 0/1 -> which 16 bd-rows
    const int half   = wv & 1;                   // 0/1 -> which 64 i-cols
    const int m      = lane & 15;                // A-frag row
    const int kg     = lane >> 4;                // lane half -> K-set select
    const int col    = lane & 15;
    const int bdBase = rowBase + mGroup * 16;
    const int iBase  = half * 64 + col;          // tiles at iBase + {0,16,32,48}

    const float*  srow = sm + (mGroup * 16 + m) * SROW;
    const __bf16* erow = ebf + (size_t)iBase * KTOT;

    f32x8 acc0 = {}, acc1 = {}, acc2 = {}, acc3 = {};

    // ---- 2-stage ping-pong pipeline: P holds chunk c, Q holds c+1 ---------
    bf16x16 aP = build_a(srow, 0, kg);
    bf16x16 p0 = load_b(erow, 0, kg, 0);
    bf16x16 p1 = load_b(erow, 0, kg, 1);
    bf16x16 p2 = load_b(erow, 0, kg, 2);
    bf16x16 p3 = load_b(erow, 0, kg, 3);

    for (int c = 0; c < NCHUNK - 2; c += 2) {
        // stage Q = chunk c+1 while P's WMMAs drain
        bf16x16 q0 = load_b(erow, c + 1, kg, 0);
        bf16x16 q1 = load_b(erow, c + 1, kg, 1);
        bf16x16 q2 = load_b(erow, c + 1, kg, 2);
        bf16x16 q3 = load_b(erow, c + 1, kg, 3);
        bf16x16 aQ = build_a(srow, c + 1, kg);
        acc0 = WMMA_BF16(aP, p0, acc0);
        acc1 = WMMA_BF16(aP, p1, acc1);
        acc2 = WMMA_BF16(aP, p2, acc2);
        acc3 = WMMA_BF16(aP, p3, acc3);
        // stage P = chunk c+2 while Q's WMMAs drain (direct writes, no copies)
        p0 = load_b(erow, c + 2, kg, 0);
        p1 = load_b(erow, c + 2, kg, 1);
        p2 = load_b(erow, c + 2, kg, 2);
        p3 = load_b(erow, c + 2, kg, 3);
        aP = build_a(srow, c + 2, kg);
        acc0 = WMMA_BF16(aQ, q0, acc0);
        acc1 = WMMA_BF16(aQ, q1, acc1);
        acc2 = WMMA_BF16(aQ, q2, acc2);
        acc3 = WMMA_BF16(aQ, q3, acc3);
    }
    {   // epilogue: chunks NCHUNK-2 (in P) and NCHUNK-1
        bf16x16 q0 = load_b(erow, NCHUNK - 1, kg, 0);
        bf16x16 q1 = load_b(erow, NCHUNK - 1, kg, 1);
        bf16x16 q2 = load_b(erow, NCHUNK - 1, kg, 2);
        bf16x16 q3 = load_b(erow, NCHUNK - 1, kg, 3);
        bf16x16 aQ = build_a(srow, NCHUNK - 1, kg);
        acc0 = WMMA_BF16(aP, p0, acc0);
        acc1 = WMMA_BF16(aP, p1, acc1);
        acc2 = WMMA_BF16(aP, p2, acc2);
        acc3 = WMMA_BF16(aP, p3, acc3);
        acc0 = WMMA_BF16(aQ, q0, acc0);
        acc1 = WMMA_BF16(aQ, q1, acc1);
        acc2 = WMMA_BF16(aQ, q2, acc2);
        acc3 = WMMA_BF16(aQ, q3, acc3);
    }

    // D layout: VGPR r -> M = r + 8*kg, N = lane&15 ; apply mask + relu
    #pragma unroll
    for (int r = 0; r < 8; ++r) {
        const int mm = r + (kg << 3);
        const int bd = bdBase + mm;
        const int dd = bd & 127;
        float v0 = acc0[r], v1 = acc1[r], v2 = acc2[r], v3 = acc3[r];
        if (iBase      == dd) v0 = 0.0f;         // mask = 1 - eye
        if (iBase + 16 == dd) v1 = 0.0f;
        if (iBase + 32 == dd) v2 = 0.0f;
        if (iBase + 48 == dd) v3 = 0.0f;
        float* orow = sout + (size_t)bd * NN;
        orow[iBase]      = fmaxf(v0, 0.0f);      // relu
        orow[iBase + 16] = fmaxf(v1, 0.0f);
        orow[iBase + 32] = fmaxf(v2, 0.0f);
        orow[iBase + 48] = fmaxf(v3, 0.0f);
    }
}

// -------------------------- 4) heads ---------------------------------------
__global__ void head_kernel(const float* __restrict__ states,
                            const float* __restrict__ rw,
                            const float* __restrict__ rb,
                            const float* __restrict__ sw,
                            const float* __restrict__ sb,
                            float* __restrict__ out) {
    __shared__ float o[128];
    const int b = blockIdx.x;
    const int tid = threadIdx.x;
    if (tid < 128) o[tid] = states[(size_t)(b * NN + tid) * NN + (NN - 1)];
    __syncthreads();
    for (int f = tid; f < INF; f += 256) {
        float acc = rb[f];
        const float* wrow = rw + (size_t)f * NN;
        #pragma unroll 8
        for (int d = 0; d < NN; ++d) acc += o[d] * wrow[d];
        out[(size_t)b * INF + f] = acc;          // recreation (flat == reshape)
    }
    if (tid == 0) {
        float acc = sb[0];
        for (int d = 0; d < NN; ++d) acc += o[d] * sw[d];
        out[(size_t)BB * INF + b] = acc;         // scores after recreation
    }
}

// ---------------------------------------------------------------------------
extern "C" void kernel_launch(void* const* d_in, const int* in_sizes, int n_in,
                              void* d_out, int out_size, void* d_ws, size_t ws_size,
                              hipStream_t stream) {
    const float* x      = (const float*)d_in[0];
    const float* ipw    = (const float*)d_in[1];
    const float* ipb    = (const float*)d_in[2];
    const float* neuron = (const float*)d_in[3];
    const float* edge   = (const float*)d_in[4];
    const float* rw     = (const float*)d_in[5];
    const float* rb     = (const float*)d_in[6];
    const float* sw     = (const float*)d_in[7];
    const float* sb     = (const float*)d_in[8];
    // grid_size=3, num_candidates=8, n_steps=2 are fixed by the reference.

    char* ws = (char*)d_ws;
    float*  statesA = (float*)(ws);
    float*  statesB = (float*)(ws + (2u << 20));
    __bf16* ebf     = (__bf16*)(ws + (4u << 20));
    float*  out     = (float*)d_out;

    proj_init_kernel<<<BB, 128, 0, stream>>>(x, ipw, ipb, neuron, statesA);
    conv_e_kernel<<<2048, 256, 0, stream>>>(edge, ebf);
    step_kernel<<<BD / 32, 128, 0, stream>>>(statesA, ebf, statesB); // step 1
    step_kernel<<<BD / 32, 128, 0, stream>>>(statesB, ebf, statesA); // step 2
    head_kernel<<<BB, 256, 0, stream>>>(statesA, rw, rb, sw, sb, out);
}
